// SimpleGAT_40845138985235
// MI455X (gfx1250) — matrix operand order repaired and checked by reading, not compiled, not run on hardware
//
#include <hip/hip_runtime.h>

typedef __attribute__((ext_vector_type(16))) __bf16 v16bf;
typedef __attribute__((ext_vector_type(8)))  __bf16 v8bf;
typedef __attribute__((ext_vector_type(8)))  float  v8f;

// ---------------------------------------------------------------- utilities
__device__ __forceinline__ void atomic_max_float(float* addr, float val) {
  // Monotonic int mapping for IEEE floats: max for >=0 via signed int max,
  // max for <0 via unsigned int min. Init value must be -inf (0xFF800000).
  if (val >= 0.0f) atomicMax((int*)addr, __float_as_int(val));
  else             atomicMin((unsigned int*)addr, __float_as_uint(val));
}

__global__ void fill_f32(float* __restrict__ p, float v, int n) {
  int t = blockIdx.x * blockDim.x + threadIdx.x;
  if (t < n) p[t] = v;
}

__global__ void cvt_f32_bf16(const float* __restrict__ in, __bf16* __restrict__ out, int n) {
  int t = blockIdx.x * blockDim.x + threadIdx.x;
  if (t < n) out[t] = (__bf16)in[t];
}

// W [K, Nc] f32 row-major  ->  Wt [Np, K] bf16 row-major (transposed, zero padded)
__global__ void transpose_w(const float* __restrict__ W, __bf16* __restrict__ Wt,
                            int K, int Nc, int Np) {
  int t = blockIdx.x * blockDim.x + threadIdx.x;
  if (t >= Np * K) return;
  int n = t / K, k = t - n * K;
  Wt[t] = (n < Nc) ? (__bf16)W[k * Nc + n] : (__bf16)0.0f;
}

// ------------------------------------------------------------- WMMA GEMM
// C[M, ncols] (+= if accum) A[M,K](bf16,row-major) * Bt[npad,K]^T (bf16).
// One wave computes one 16x16 tile via v_wmma_f32_16x16x32_bf16.
// 2 waves / block along M; grid = (ceil(Mtiles/2), npad/16).
__global__ void gemm_wmma_bf16(const __bf16* __restrict__ A, int lda,
                               const __bf16* __restrict__ Bt, int ldb,
                               float* __restrict__ Cf, __bf16* __restrict__ Cbf,
                               const float* __restrict__ bias,
                               int M, int K, int ldc, int ncols, int accum) {
  const int lane   = threadIdx.x & 31;
  const int wave   = threadIdx.x >> 5;
  const int tile_m = blockIdx.x * 2 + wave;
  const int tile_n = blockIdx.y;
  const int idx    = lane & 15;
  const int half   = lane >> 4;

  int row = tile_m * 16 + idx;
  if (row >= M) row = M - 1;              // clamp loads for ragged tail
  const int ncol = tile_n * 16 + idx;     // output column / Bt row (padded: always valid)
  const int kb   = half * 8;              // A-frag K sub-offset (ISA 16-bit A layout)

  v8f c = {};
  if (accum) {
#pragma unroll
    for (int r = 0; r < 8; ++r) {
      int m = tile_m * 16 + r + 8 * half;
      c[r] = (m < M && ncol < ncols) ? Cf[(size_t)m * ldc + ncol] : 0.0f;
    }
  }

  const __bf16* ap = A  + (size_t)row  * lda + kb;        // chunks [kb,kb+8) & [kb+16,kb+24)
  const __bf16* bp = Bt + (size_t)ncol * ldb + kb;        // chunks [kb,kb+8) & [kb+16,kb+24)
  for (int k0 = 0; k0 < K; k0 += 32) {
    v8bf a0 = *(const v8bf*)(ap + k0);
    v8bf a1 = *(const v8bf*)(ap + k0 + 16);
    v8bf b0 = *(const v8bf*)(bp + k0);
    v8bf b1 = *(const v8bf*)(bp + k0 + 16);
    v16bf a = __builtin_shufflevector(a0, a1, 0,1,2,3,4,5,6,7,8,9,10,11,12,13,14,15);
    v16bf b = __builtin_shufflevector(b0, b1, 0,1,2,3,4,5,6,7,8,9,10,11,12,13,14,15);
    c = __builtin_amdgcn_wmma_f32_16x16x32_bf16(false, a, false, b, (short)0, c, false, false);
  }

  const float bv = (bias != nullptr && ncol < ncols) ? bias[ncol] : 0.0f;
#pragma unroll
  for (int r = 0; r < 8; ++r) {
    int m = tile_m * 16 + r + 8 * half;
    if (m < M && ncol < ncols) {
      float v = c[r] + bv;
      if (Cf)  Cf [(size_t)m * ldc + ncol] = v;
      if (Cbf) Cbf[(size_t)m * ldc + ncol] = (__bf16)v;
    }
  }
}

// ------------------------------------------------------------- edge passes
// Pass A: score[e,h] = att_h . leakyrelu(xl[src] + xr[dst]);  m[dst,h] = max(...)
template <int H>
__global__ void edge_score_kernel(const int* __restrict__ ei, int E, int N,
                                  const __bf16* __restrict__ xl,
                                  const __bf16* __restrict__ xr,
                                  const float* __restrict__ att,
                                  float* __restrict__ score,
                                  float* __restrict__ mmax) {
  int t = blockIdx.x * blockDim.x + threadIdx.x;
  const int Etot = E + N;
  if (t >= Etot * H) return;
  const int e = t / H, h = t - e * H;
  const int s = (e < E) ? ei[e]     : (e - E);   // self loops appended
  const int d = (e < E) ? ei[E + e] : (e - E);
  const __bf16* pl = xl + (size_t)s * (64 * H) + h * 64;
  const __bf16* pr = xr + (size_t)d * (64 * H) + h * 64;
  const float*  a  = att + h * 64;
  float acc = 0.0f;
#pragma unroll
  for (int c0 = 0; c0 < 64; c0 += 8) {
    v8bf l = *(const v8bf*)(pl + c0);
    v8bf r = *(const v8bf*)(pr + c0);
#pragma unroll
    for (int i = 0; i < 8; ++i) {
      float v = (float)l[i] + (float)r[i];
      v = (v >= 0.0f) ? v : 0.2f * v;
      acc += a[c0 + i] * v;
    }
  }
  score[t] = acc;
  atomic_max_float(&mmax[(size_t)d * H + h], acc);
}

// Pass B: ex = exp(score - m[dst]); denom[dst] += ex; acc[dst] += ex * xl[src]
template <int H>
__global__ void edge_accum_kernel(const int* __restrict__ ei, int E, int N,
                                  const __bf16* __restrict__ xl,
                                  const float* __restrict__ score,
                                  const float* __restrict__ mmax,
                                  float* __restrict__ denom,
                                  float* __restrict__ acc) {
  int t = blockIdx.x * blockDim.x + threadIdx.x;
  const int Etot = E + N;
  if (t >= Etot * H) return;
  const int e = t / H, h = t - e * H;
  const int s = (e < E) ? ei[e]     : (e - E);
  const int d = (e < E) ? ei[E + e] : (e - E);
  const float ex = expf(score[t] - mmax[(size_t)d * H + h]);
  atomicAdd(&denom[(size_t)d * H + h], ex);
  const __bf16* pl = xl  + (size_t)s * (64 * H) + h * 64;
  float*        pa = acc + (size_t)d * (64 * H) + h * 64;
#pragma unroll
  for (int c0 = 0; c0 < 64; c0 += 8) {
    v8bf l = *(const v8bf*)(pl + c0);
#pragma unroll
    for (int i = 0; i < 8; ++i) atomicAdd(&pa[c0 + i], ex * (float)l[i]);
  }
}

// Pass C: out = acc / denom + bias  (+ ELU for layer 1), emit bf16
template <int H, bool ELU>
__global__ void finalize_kernel(const float* __restrict__ acc,
                                const float* __restrict__ denom,
                                const float* __restrict__ bias,
                                __bf16* __restrict__ outbf, int N) {
  int t = blockIdx.x * blockDim.x + threadIdx.x;
  if (t >= N * 64 * H) return;
  int node = t / (64 * H);
  int ch   = t - node * (64 * H);
  int h    = ch >> 6;
  float v = acc[t] / denom[(size_t)node * H + h] + bias[ch];
  if (ELU) v = (v > 0.0f) ? v : expm1f(v);
  outbf[t] = (__bf16)v;
}

// --------------------------------------------------------------- launcher
extern "C" void kernel_launch(void* const* d_in, const int* in_sizes, int n_in,
                              void* d_out, int out_size, void* d_ws, size_t ws_size,
                              hipStream_t stream) {
  const float* x    = (const float*)d_in[0];
  const int*   ei   = (const int*)  d_in[1];
  const float* Wl1  = (const float*)d_in[2];
  const float* Wr1  = (const float*)d_in[3];
  const float* att1 = (const float*)d_in[4];
  const float* b1   = (const float*)d_in[5];
  const float* Wl2  = (const float*)d_in[6];
  const float* Wr2  = (const float*)d_in[7];
  const float* att2 = (const float*)d_in[8];
  const float* b2   = (const float*)d_in[9];
  const float* Wc   = (const float*)d_in[10];
  const float* bc   = (const float*)d_in[11];
  float* out = (float*)d_out;

  const int IN = 128, OUT = 40, OUTP = 48;
  const int N    = in_sizes[0] / IN;
  const int E    = in_sizes[1] / 2;
  const int Etot = E + N;

  // ---- workspace carve-out (256B aligned) ----
  char* wsb = (char*)d_ws; size_t off = 0;
  auto take = [&](size_t bytes) -> void* {
    void* p = wsb + off;
    off = (off + bytes + 255) & ~(size_t)255;
    return p;
  };
  __bf16* xbf  = (__bf16*)take((size_t)N * 128 * 2);
  __bf16* Wl1t = (__bf16*)take(128 * 128 * 2);
  __bf16* Wr1t = (__bf16*)take(128 * 128 * 2);
  __bf16* Wl2t = (__bf16*)take(64 * 128 * 2);
  __bf16* Wr2t = (__bf16*)take(64 * 128 * 2);
  __bf16* Wct  = (__bf16*)take((size_t)OUTP * 192 * 2);
  __bf16* xl1  = (__bf16*)take((size_t)N * 128 * 2);
  __bf16* xr1  = (__bf16*)take((size_t)N * 128 * 2);
  float*  sc1  = (float*) take((size_t)Etot * 2 * 4);
  float*  m1   = (float*) take((size_t)N * 2 * 4);
  float*  dn1  = (float*) take((size_t)N * 2 * 4);
  float*  acc1 = (float*) take((size_t)N * 128 * 4);
  __bf16* h1   = (__bf16*)take((size_t)N * 128 * 2);
  __bf16* xl2  = (__bf16*)take((size_t)N * 64 * 2);
  __bf16* xr2  = (__bf16*)take((size_t)N * 64 * 2);
  float*  sc2  = (float*) take((size_t)Etot * 4);
  float*  m2   = (float*) take((size_t)N * 4);
  float*  dn2  = (float*) take((size_t)N * 4);
  float*  acc2 = (float*) take((size_t)N * 64 * 4);
  __bf16* h2   = (__bf16*)take((size_t)N * 64 * 2);
  (void)ws_size; (void)n_in; (void)out_size;

  auto cdiv = [](int a, int b) { return (a + b - 1) / b; };
  auto gemm = [&](const __bf16* A, int lda, const __bf16* Bt, int ldb,
                  float* Cf, __bf16* Cbf, const float* bias,
                  int M, int K, int ldc, int ncols, int npad, int accum) {
    dim3 g(cdiv(cdiv(M, 16), 2), npad / 16);
    gemm_wmma_bf16<<<g, 64, 0, stream>>>(A, lda, Bt, ldb, Cf, Cbf, bias, M, K, ldc, ncols, accum);
  };

  // ---- prep: bf16 conversions & weight transposes ----
  cvt_f32_bf16<<<cdiv(N * 128, 256), 256, 0, stream>>>(x, xbf, N * 128);
  transpose_w<<<cdiv(128 * 128, 256), 256, 0, stream>>>(Wl1, Wl1t, 128, 128, 128);
  transpose_w<<<cdiv(128 * 128, 256), 256, 0, stream>>>(Wr1, Wr1t, 128, 128, 128);
  transpose_w<<<cdiv(64 * 128, 256), 256, 0, stream>>>(Wl2, Wl2t, 128, 64, 64);
  transpose_w<<<cdiv(64 * 128, 256), 256, 0, stream>>>(Wr2, Wr2t, 128, 64, 64);
  transpose_w<<<cdiv(OUTP * 192, 256), 256, 0, stream>>>(Wc, Wct, 192, OUT, OUTP);

  // ---- layer 1: GATv2(128 -> 64, heads=2, concat) ----
  fill_f32<<<cdiv(N * 2, 256), 256, 0, stream>>>(m1, -INFINITY, N * 2);
  fill_f32<<<cdiv(N * 2, 256), 256, 0, stream>>>(dn1, 0.0f, N * 2);
  fill_f32<<<cdiv(N * 128, 256), 256, 0, stream>>>(acc1, 0.0f, N * 128);
  gemm(xbf, 128, Wl1t, 128, nullptr, xl1, nullptr, N, 128, 128, 128, 128, 0);
  gemm(xbf, 128, Wr1t, 128, nullptr, xr1, nullptr, N, 128, 128, 128, 128, 0);
  edge_score_kernel<2><<<cdiv(Etot * 2, 256), 256, 0, stream>>>(ei, E, N, xl1, xr1, att1, sc1, m1);
  edge_accum_kernel<2><<<cdiv(Etot * 2, 256), 256, 0, stream>>>(ei, E, N, xl1, sc1, m1, dn1, acc1);
  finalize_kernel<2, true><<<cdiv(N * 128, 256), 256, 0, stream>>>(acc1, dn1, b1, h1, N);

  // ---- layer 2: GATv2(128 -> 64, heads=1) ----
  fill_f32<<<cdiv(N, 256), 256, 0, stream>>>(m2, -INFINITY, N);
  fill_f32<<<cdiv(N, 256), 256, 0, stream>>>(dn2, 0.0f, N);
  fill_f32<<<cdiv(N * 64, 256), 256, 0, stream>>>(acc2, 0.0f, N * 64);
  gemm(h1, 128, Wl2t, 128, nullptr, xl2, nullptr, N, 128, 64, 64, 64, 0);
  gemm(h1, 128, Wr2t, 128, nullptr, xr2, nullptr, N, 128, 64, 64, 64, 0);
  edge_score_kernel<1><<<cdiv(Etot, 256), 256, 0, stream>>>(ei, E, N, xl2, xr2, att2, sc2, m2);
  edge_accum_kernel<1><<<cdiv(Etot, 256), 256, 0, stream>>>(ei, E, N, xl2, sc2, m2, dn2, acc2);
  finalize_kernel<1, false><<<cdiv(N * 64, 256), 256, 0, stream>>>(acc2, dn2, b2, h2, N);

  // ---- classifier: [h2 | x] @ Wc + bc, split over K (no concat materialized) ----
  gemm(h2, 64, Wct, 192, out, nullptr, bc, N, 64, OUT, OUT, OUTP, 0);        // h2 @ Wc[0:64]
  gemm(xbf, 128, Wct + 64, 192, out, nullptr, nullptr, N, 128, OUT, OUT, OUTP, 1); // x @ Wc[64:192]
}
